// CausalSelfAttention_31344671326806
// MI455X (gfx1250) — matrix operand order repaired and checked by reading, compiled, not run on hardware
//
#include <hip/hip_runtime.h>
#include <hip/hip_bf16.h>

typedef __bf16 bf16_t;
typedef __attribute__((ext_vector_type(16))) __bf16 v16bf;
typedef __attribute__((ext_vector_type(8)))  __bf16 v8bf;
typedef __attribute__((ext_vector_type(4)))  __bf16 v4bf;
typedef __attribute__((ext_vector_type(8)))  float  v8f;
typedef __attribute__((ext_vector_type(4)))  float  v4f;
typedef int gv4i __attribute__((vector_size(16)));   // matches builtin's V4i

#define DIM     2048
#define NHEADS  16
#define NKV     4
#define HD      128
#define KV_DIM  512
#define SEQ     2048
#define BATCH   4
#define QKV_N   3072
#define NTOK    (BATCH * SEQ)

// ---------------------------------------------------------------------------
// CDNA5 async global->LDS copies (ASYNCcnt path), with sync fallback.
// ---------------------------------------------------------------------------
#if defined(__has_builtin)
# if __has_builtin(__builtin_amdgcn_global_load_async_to_lds_b128)
#  define HAVE_ASYNC_LDS 1
# endif
#endif
#ifndef HAVE_ASYNC_LDS
# define HAVE_ASYNC_LDS 0
#endif

__device__ __forceinline__ void copy_b128_to_lds(bf16_t* ldsp, const bf16_t* gp) {
#if HAVE_ASYNC_LDS
  __builtin_amdgcn_global_load_async_to_lds_b128(
      (__attribute__((address_space(1))) gv4i*)gp,
      (__attribute__((address_space(3))) gv4i*)ldsp,
      /*offset=*/0, /*cpol=*/0);
#else
  *(v8bf*)ldsp = *(const v8bf*)gp;
#endif
}

#if HAVE_ASYNC_LDS
# if __has_builtin(__builtin_amdgcn_s_wait_asynccnt)
#  define WAIT_ASYNC(n) __builtin_amdgcn_s_wait_asynccnt(n)
# else
#  define WAIT_ASYNC(n) asm volatile("s_wait_asynccnt %0" :: "n"(n) : "memory")
# endif
#else
# define WAIT_ASYNC(n) ((void)0)
#endif

// ---------------------------------------------------------------------------
// WMMA helpers (CDNA5 v_wmma_f32_16x16x32_bf16).
// A-matrix 16x32 bf16 lane layout (ISA 7.12.2): lanes 0-15 row M=lane hold
// K=0..7 / K=16..23; lanes 16-31 hold K=8..15 / K=24..31.
// B-matrix 32x16: lanes 0-15 col N=lane hold K=0..7 / K=16..23; lanes 16-31 K=8..15 / K=24..31.
// ---------------------------------------------------------------------------
__device__ __forceinline__ v16bf frag_cat(v8bf lo, v8bf hi) {
  return __builtin_shufflevector(lo, hi, 0, 1, 2, 3, 4, 5, 6, 7,
                                 8, 9, 10, 11, 12, 13, 14, 15);
}

__device__ __forceinline__ v16bf load_fragA(const bf16_t* p, int half) {
  v8bf lo = *(const v8bf*)(p + (half ? 8 : 0));
  v8bf hi = *(const v8bf*)(p + (half ? 24 : 16));
  return frag_cat(lo, hi);
}

__device__ __forceinline__ v16bf load_fragB(const bf16_t* p, int half) {
  v8bf lo = *(const v8bf*)(p + (half ? 8 : 0));
  v8bf hi = *(const v8bf*)(p + (half ? 24 : 16));
  return frag_cat(lo, hi);
}

__device__ __forceinline__ v8f wmma_bf16(v16bf a, v16bf b, v8f c) {
  return __builtin_amdgcn_wmma_f32_16x16x32_bf16(
      false, a, false, b, (short)0, c, false, false);
}

// ---------------------------------------------------------------------------
// fp32 -> bf16 conversion, 4 elements / thread
// ---------------------------------------------------------------------------
__global__ __launch_bounds__(256) void cvt_f32_bf16(
    const float* __restrict__ in, bf16_t* __restrict__ out, int n4) {
  int i = blockIdx.x * 256 + threadIdx.x;
  if (i < n4) {
    v4f v = *(const v4f*)(in + (size_t)i * 4);
    v4bf o;
    o.x = (__bf16)v.x; o.y = (__bf16)v.y; o.z = (__bf16)v.z; o.w = (__bf16)v.w;
    *(v4bf*)(out + (size_t)i * 4) = o;
  }
}

// ---------------------------------------------------------------------------
// fp32 W[K][N] -> bf16 WT[N][K]  (tiled 32x32 LDS transpose, coalesced)
// grid: (N/32, K/32)
// ---------------------------------------------------------------------------
__global__ __launch_bounds__(256) void wtrans_kernel(
    const float* __restrict__ W, bf16_t* __restrict__ WT, int K, int N) {
  __shared__ float t[32][33];
  const int n0 = blockIdx.x * 32, k0 = blockIdx.y * 32;
  const int tid = threadIdx.x;
#pragma unroll
  for (int i = tid; i < 1024; i += 256) {
    int r = i >> 5, c = i & 31;                       // r = k, c = n
    t[r][c] = W[(size_t)(k0 + r) * N + n0 + c];
  }
  __syncthreads();
#pragma unroll
  for (int i = tid; i < 1024; i += 256) {
    int r = i >> 5, c = i & 31;                       // r = n, c = k
    WT[(size_t)(n0 + r) * K + k0 + c] = (bf16_t)t[c][r];
  }
}

// ---------------------------------------------------------------------------
// V slice of qkv (f32) -> transposed bf16 Vt [b, kv, hd, seq]
// grid: (SEQ/32, HD/32, BATCH*NKV)
// ---------------------------------------------------------------------------
__global__ __launch_bounds__(256) void vtrans_kernel(
    const float* __restrict__ qkvf, bf16_t* __restrict__ vtb) {
  __shared__ float t[32][33];
  const int s0 = blockIdx.x * 32;
  const int d0 = blockIdx.y * 32;
  const int bk = blockIdx.z;                          // b*NKV + kh
  const int b = bk >> 2, kh = bk & 3;
  const int tid = threadIdx.x;
#pragma unroll
  for (int i = tid; i < 1024; i += 256) {
    int r = i >> 5, c = i & 31;                       // r = s, c = d
    t[r][c] = qkvf[(size_t)(b * SEQ + s0 + r) * QKV_N +
                   DIM + KV_DIM + kh * HD + d0 + c];
  }
  __syncthreads();
#pragma unroll
  for (int i = tid; i < 1024; i += 256) {
    int r = i >> 5, c = i & 31;                       // r = d, c = s
    vtb[((size_t)bk * HD + d0 + r) * SEQ + s0 + c] = (bf16_t)t[c][r];
  }
}

// ---------------------------------------------------------------------------
// bf16 GEMM: C[M,N] (f32) = A[M,K] @ BT[N,K]^T.  128x128 block, BK=64,
// double-buffered LDS fed by async global->LDS b128 copies.
// 8 waves (4 row-groups x 2 col-groups), each wave 32x64 = 2x4 WMMA tiles.
// M,N multiples of 128; K multiple of 64.
// ---------------------------------------------------------------------------
__global__ __launch_bounds__(256) void gemm_bf16_kernel(
    const bf16_t* __restrict__ A, const bf16_t* __restrict__ BT,
    float* __restrict__ C, int M, int N, int K) {
  __shared__ bf16_t As[2][128][64];
  __shared__ bf16_t Bs[2][128][64];

  const int tid  = threadIdx.x;
  const int lane = tid & 31;
  const int wave = tid >> 5;
  const int wm   = wave & 3;       // row group (32 rows each)
  const int wn   = wave >> 2;      // col group (64 cols each)
  const int half = lane >> 4;
  const int l16  = lane & 15;
  const int row0 = blockIdx.y * 128;
  const int col0 = blockIdx.x * 128;
  const bf16_t* Ab = A  + (size_t)row0 * K;
  const bf16_t* Bb = BT + (size_t)col0 * K;

  const v8f zacc = {};
  v8f acc[2][4];
#pragma unroll
  for (int i = 0; i < 2; ++i)
#pragma unroll
    for (int j = 0; j < 4; ++j) acc[i][j] = zacc;

  auto issue_stage = [&](int stg) {
    const int k0  = stg * 64;
    const int buf = stg & 1;
#pragma unroll
    for (int c = tid; c < 1024; c += 256) {           // A tile 128x64
      int r = c >> 3, cc = c & 7;
      copy_b128_to_lds(&As[buf][r][cc * 8], &Ab[(size_t)r * K + k0 + cc * 8]);
    }
#pragma unroll
    for (int c = tid; c < 1024; c += 256) {           // B tile 128x64 (N-major)
      int r = c >> 3, cc = c & 7;
      copy_b128_to_lds(&Bs[buf][r][cc * 8], &Bb[(size_t)r * K + k0 + cc * 8]);
    }
  };

  const int nStg = K >> 6;
  issue_stage(0);
  for (int s = 0; s < nStg; ++s) {
    const int buf = s & 1;
    if (s + 1 < nStg) { issue_stage(s + 1); WAIT_ASYNC(8); }
    else              { WAIT_ASYNC(0); }
    __syncthreads();

#pragma unroll
    for (int ks = 0; ks < 2; ++ks) {
#pragma unroll
      for (int mt = 0; mt < 2; ++mt) {
        v16bf a = load_fragA(&As[buf][wm * 32 + mt * 16 + l16][ks * 32], half);
#pragma unroll
        for (int nt = 0; nt < 4; ++nt) {
          v16bf b = load_fragB(&Bs[buf][wn * 64 + nt * 16 + l16][ks * 32], half);
          acc[mt][nt] = wmma_bf16(a, b, acc[mt][nt]);
        }
      }
    }
    __syncthreads();
  }

  // D layout: VGPR r -> row r (lanes 0-15) / row 8+r (lanes 16-31), col = lane%16
#pragma unroll
  for (int mt = 0; mt < 2; ++mt) {
    int mbase = row0 + wm * 32 + mt * 16 + half * 8;
#pragma unroll
    for (int nt = 0; nt < 4; ++nt) {
      int ncol = col0 + wn * 64 + nt * 16 + l16;
#pragma unroll
      for (int r = 0; r < 8; ++r)
        C[(size_t)(mbase + r) * N + ncol] = acc[mt][nt][r];
    }
  }
}

// ---------------------------------------------------------------------------
// Fused per-head RMSNorm + RoPE; writes head-major bf16 Q/K.
// One 128-thread block per (token, unit): units 0-15 = Q heads, 16-19 = K.
// ---------------------------------------------------------------------------
__global__ __launch_bounds__(128) void normrope_kernel(
    const float* __restrict__ qkv, bf16_t* __restrict__ qb,
    bf16_t* __restrict__ kb) {
  const int unit  = blockIdx.x % 20;
  const int token = blockIdx.x / 20;
  const int b = token >> 11;        // / SEQ
  const int s = token & 2047;       // % SEQ
  const int d = threadIdx.x;
  const float* base = qkv + (size_t)token * QKV_N;

  int off;
  bf16_t* dst;
  if (unit < 16) {
    off = unit * HD;
    dst = qb + (((size_t)(b * NHEADS + unit)) * SEQ + s) * HD;
  } else {
    int kh = unit - 16;
    off = DIM + kh * HD;
    dst = kb + (((size_t)(b * NKV + kh)) * SEQ + s) * HD;
  }

  float x = base[off + d];
  float ss = x * x;
#pragma unroll
  for (int m = 16; m >= 1; m >>= 1) ss += __shfl_xor(ss, m, 32);
  __shared__ float red[4];
  __shared__ float xs[128];
  if ((d & 31) == 0) red[d >> 5] = ss;
  __syncthreads();
  float tot = red[0] + red[1] + red[2] + red[3];
  float xn = x * rsqrtf(tot * (1.0f / 128.0f) + 1.1920928955078125e-07f);
  xs[d] = xn;
  __syncthreads();

  int j = d & 63;
  float inv_freq = __expf(-(float)j * (9.210340371976184f / 64.0f)); // 10000^(-j/64)
  float ang = (float)s * inv_freq;
  float sn, cs;
  __sincosf(ang, &sn, &cs);
  float out = (d < 64) ? (xn * cs + xs[d + 64] * sn)
                       : (xn * cs - xs[d - 64] * sn);
  dst[d] = (bf16_t)out;
}

// ---------------------------------------------------------------------------
// Causal flash attention.  Block = (b, head, 128-row Q tile), 8 waves; each
// wave owns 16 Q rows.  Q fragments live in registers; K (row-major) and
// pre-transposed V tiles are staged via async global->LDS b128 copies so all
// WMMA fragment reads are contiguous ds_load_b128.  Online softmax in f32 on
// the WMMA D layout; P staged via LDS bf16 (wave-local rows, no barrier).
// ---------------------------------------------------------------------------
__global__ __launch_bounds__(256) void attn_kernel(
    const bf16_t* __restrict__ qb, const bf16_t* __restrict__ kb,
    const bf16_t* __restrict__ vtb, bf16_t* __restrict__ yb) {
  __shared__ bf16_t Ks[128][128];  // [key][hd]
  __shared__ bf16_t Vt[128][128];  // [hd][key]
  __shared__ bf16_t Ps[128][128];  // [q-row][key]

  const int tid  = threadIdx.x;
  const int lane = tid & 31;
  const int wave = tid >> 5;
  const int half = lane >> 4;
  const int l16  = lane & 15;
  const int qt = blockIdx.x;
  const int h  = blockIdx.y;
  const int b  = blockIdx.z;
  const int kh = h >> 2;                    // GQA: 4 Q heads per KV head

  // Q fragments: 4 K-chunks (head dim 128 = 4 x 32) for this wave's 16 rows
  const bf16_t* qrow =
      qb + (((size_t)(b * NHEADS + h)) * SEQ + qt * 128 + wave * 16 + l16) * HD;
  v16bf qf[4];
#pragma unroll
  for (int ks = 0; ks < 4; ++ks) qf[ks] = load_fragA(qrow + ks * 32, half);

  const bf16_t* kbase  = kb  + ((size_t)(b * NKV + kh)) * SEQ * HD;   // [s][d]
  const bf16_t* vtbase = vtb + ((size_t)(b * NKV + kh)) * HD * SEQ;   // [d][s]

  const v8f zacc = {};
  v8f O[8];
#pragma unroll
  for (int i = 0; i < 8; ++i) O[i] = zacc;
  float mrow[8], lrow[8];
#pragma unroll
  for (int r = 0; r < 8; ++r) { mrow[r] = -1e30f; lrow[r] = 0.0f; }

  const int qrow_base = qt * 128 + wave * 16 + half * 8;
  const float scale = 0.08838834764831845f;  // 1/sqrt(128)

  for (int j = 0; j <= qt; ++j) {
    __syncthreads();                       // prior compute done before overwrite
#pragma unroll
    for (int c = tid; c < 2048; c += 256) {   // K tile 128x128
      int r = c >> 4, cc = c & 15;
      copy_b128_to_lds(&Ks[r][cc * 8],
                       &kbase[(size_t)(j * 128 + r) * HD + cc * 8]);
    }
#pragma unroll
    for (int c = tid; c < 2048; c += 256) {   // V tile (already transposed)
      int r = c >> 4, cc = c & 15;
      copy_b128_to_lds(&Vt[r][cc * 8],
                       &vtbase[(size_t)r * SEQ + j * 128 + cc * 8]);
    }
    WAIT_ASYNC(0);
    __syncthreads();

    // S = Q @ K^T : 8 key-subtiles x 4 k-steps
    v8f sacc[8];
#pragma unroll
    for (int nt = 0; nt < 8; ++nt) sacc[nt] = zacc;
#pragma unroll
    for (int ks = 0; ks < 4; ++ks) {
#pragma unroll
      for (int nt = 0; nt < 8; ++nt) {
        v16bf bf = load_fragB(&Ks[nt * 16 + l16][ks * 32], half);
        sacc[nt] = wmma_bf16(qf[ks], bf, sacc[nt]);
      }
    }

    // scale + causal mask + per-row max
    const bool lastT = (j == qt);
    float tmax[8];
#pragma unroll
    for (int r = 0; r < 8; ++r) tmax[r] = -1e30f;
#pragma unroll
    for (int nt = 0; nt < 8; ++nt) {
      int key = j * 128 + nt * 16 + l16;
#pragma unroll
      for (int r = 0; r < 8; ++r) {
        float sv = sacc[nt][r] * scale;
        if (lastT && key > qrow_base + r) sv = -1e30f;
        sacc[nt][r] = sv;
        tmax[r] = fmaxf(tmax[r], sv);
      }
    }
#pragma unroll
    for (int r = 0; r < 8; ++r) {
#pragma unroll
      for (int m = 8; m >= 1; m >>= 1)
        tmax[r] = fmaxf(tmax[r], __shfl_xor(tmax[r], m, 16));
    }

    float alpha[8];
#pragma unroll
    for (int r = 0; r < 8; ++r) {
      float mn = fmaxf(mrow[r], tmax[r]);
      alpha[r] = __expf(mrow[r] - mn);
      mrow[r] = mn;
    }
    float lsum[8];
#pragma unroll
    for (int r = 0; r < 8; ++r) lsum[r] = 0.0f;
#pragma unroll
    for (int nt = 0; nt < 8; ++nt) {
#pragma unroll
      for (int r = 0; r < 8; ++r) {
        float p = __expf(sacc[nt][r] - mrow[r]);
        sacc[nt][r] = p;
        lsum[r] += p;
      }
    }
#pragma unroll
    for (int r = 0; r < 8; ++r) {
#pragma unroll
      for (int m = 8; m >= 1; m >>= 1) lsum[r] += __shfl_xor(lsum[r], m, 16);
      lrow[r] = lrow[r] * alpha[r] + lsum[r];
    }
#pragma unroll
    for (int dt = 0; dt < 8; ++dt)
#pragma unroll
      for (int r = 0; r < 8; ++r) O[dt][r] *= alpha[r];

    // stage P (D layout -> row-major bf16 A layout); rows are wave-local
#pragma unroll
    for (int nt = 0; nt < 8; ++nt) {
      int col = nt * 16 + l16;
#pragma unroll
      for (int r = 0; r < 8; ++r)
        Ps[wave * 16 + half * 8 + r][col] = (bf16_t)sacc[nt][r];
    }

    // O += P @ V : 8 hd-subtiles x 4 key-steps
#pragma unroll
    for (int ks = 0; ks < 4; ++ks) {
      v16bf a = load_fragA(&Ps[wave * 16 + l16][ks * 32], half);
#pragma unroll
      for (int dt = 0; dt < 8; ++dt) {
        v16bf bf = load_fragB(&Vt[dt * 16 + l16][ks * 32], half);
        O[dt] = wmma_bf16(a, bf, O[dt]);
      }
    }
  }

  // normalize and write y as [B, S, DIM] bf16
#pragma unroll
  for (int r = 0; r < 8; ++r) lrow[r] = 1.0f / lrow[r];
  const int srow = qt * 128 + wave * 16 + half * 8;
#pragma unroll
  for (int dt = 0; dt < 8; ++dt) {
    int dcol = h * HD + dt * 16 + l16;
#pragma unroll
    for (int r = 0; r < 8; ++r)
      yb[(((size_t)b * SEQ) + srow + r) * DIM + dcol] =
          (bf16_t)(O[dt][r] * lrow[r]);
  }
}

// ---------------------------------------------------------------------------
// Host launcher
// ---------------------------------------------------------------------------
extern "C" void kernel_launch(void* const* d_in, const int* in_sizes, int n_in,
                              void* d_out, int out_size, void* d_ws,
                              size_t ws_size, hipStream_t stream) {
  const float* x    = (const float*)d_in[0];
  const float* Wqkv = (const float*)d_in[1];
  const float* Wout = (const float*)d_in[2];
  float* out = (float*)d_out;

  char* ws = (char*)d_ws;
  size_t off = 0;
  bf16_t* xb    = (bf16_t*)(ws + off); off += (size_t)NTOK * DIM * 2;
  bf16_t* wqkvT = (bf16_t*)(ws + off); off += (size_t)DIM * QKV_N * 2;
  bf16_t* woutT = (bf16_t*)(ws + off); off += (size_t)DIM * DIM * 2;
  float*  qkvf  = (float*) (ws + off); off += (size_t)NTOK * QKV_N * 4;
  bf16_t* qbuf  = (bf16_t*)(ws + off); off += (size_t)NTOK * DIM * 2;
  bf16_t* kbuf  = (bf16_t*)(ws + off); off += (size_t)NTOK * KV_DIM * 2;
  bf16_t* vtb   = (bf16_t*)(ws + off); off += (size_t)NTOK * KV_DIM * 2;
  bf16_t* ybuf  = (bf16_t*)(ws + off); off += (size_t)NTOK * DIM * 2;

  // 1) activations -> bf16; weights -> transposed bf16
  {
    int n4 = NTOK * DIM / 4;
    cvt_f32_bf16<<<(n4 + 255) / 256, 256, 0, stream>>>(x, xb, n4);
  }
  wtrans_kernel<<<dim3(QKV_N / 32, DIM / 32), 256, 0, stream>>>(
      Wqkv, wqkvT, DIM, QKV_N);
  wtrans_kernel<<<dim3(DIM / 32, DIM / 32), 256, 0, stream>>>(
      Wout, woutT, DIM, DIM);

  // 2) QKV projection: [8192,2048] @ [2048,3072] -> f32
  gemm_bf16_kernel<<<dim3(QKV_N / 128, NTOK / 128), 256, 0, stream>>>(
      xb, wqkvT, qkvf, NTOK, QKV_N, DIM);

  // 3) RMSNorm + RoPE for Q/K; V -> transposed bf16
  normrope_kernel<<<NTOK * 20, 128, 0, stream>>>(qkvf, qbuf, kbuf);
  vtrans_kernel<<<dim3(SEQ / 32, HD / 32, BATCH * NKV), 256, 0, stream>>>(
      qkvf, vtb);

  // 4) causal flash attention -> y [B,S,DIM] bf16
  attn_kernel<<<dim3(SEQ / 128, NHEADS, BATCH), 256, 0, stream>>>(
      qbuf, kbuf, vtb, ybuf);

  // 5) output projection: [8192,2048] @ [2048,2048] -> f32 out
  gemm_bf16_kernel<<<dim3(DIM / 128, NTOK / 128), 256, 0, stream>>>(
      ybuf, woutT, out, NTOK, DIM, DIM);
}
